// AttnGINProtEmb_35390530519809
// MI455X (gfx1250) — matrix operand order repaired and checked by reading, not compiled, not run on hardware
//
#include <hip/hip_runtime.h>

// ---------------------------------------------------------------------------
// AttnGINProtEmb for MI455X (gfx1250, wave32, WMMA bf16 16x16x32)
//
// Shapes: B=256, N=64, L=1000, D=128, F_IN=78, E_PER_G=256, VOCAB=26
//   NTOT = 16384 nodes, ETOT = 65536 edges
//
// Key algebraic optimizations (exact, exploiting VOCAB=26):
//  * attention over L=1000 tokens == count-weighted softmax over 26 logits
//  * xt = emb_seq @ w_xt == gather-sum of a [1000,26,128] pos table that is
//    produced by a batched 26x128x128 WMMA GEMM and stays L2-resident (16MB).
// GEMM data movement:
//  * A fragments: global -> registers directly (per-lane contiguous K runs).
//  * B tile: LDS in fragment-major layout (2 x ds_load_b128 per fragment),
//    double-buffered so staging overlaps WMMA; one barrier per K-step.
//  * fp32 -> bf16 via hardware v_cvt_pk_bf16_f32 (pairs), not bit twiddling.
//  * Branchless bounds: OOB M-rows / N-cols only feed accumulator lanes the
//    epilogue never writes; addresses clamped, only K-tail zero-selected.
// ---------------------------------------------------------------------------

typedef __attribute__((ext_vector_type(16))) __bf16 v16bf;
typedef __attribute__((ext_vector_type(8)))  __bf16 v8bf;
typedef __attribute__((ext_vector_type(4)))  __bf16 v4bf;
typedef __attribute__((ext_vector_type(2)))  __bf16 v2bf;
typedef __attribute__((ext_vector_type(8)))  float  v8f;
typedef __attribute__((ext_vector_type(4)))  float  v4f;

// pair conversion: 2 x f32 -> packed 2 x bf16 (one VALU op on CDNA5)
static __device__ __forceinline__ v2bf pk2(float a, float b) {
#if __has_builtin(__builtin_amdgcn_cvt_pk_bf16_f32)
  return __builtin_amdgcn_cvt_pk_bf16_f32(a, b);
#else
  v2bf r;
  r[0] = (__bf16)a;
  r[1] = (__bf16)b;
  return r;
#endif
}

static __device__ __forceinline__ v4bf cvt4(v4f q) {
  const v2bf a = pk2(q[0], q[1]);
  const v2bf b = pk2(q[2], q[3]);
  v4bf r;
  r[0] = a[0]; r[1] = a[1]; r[2] = b[0]; r[3] = b[1];
  return r;
}

// ---------------------------------------------------------------------------
// Generic tiled WMMA GEMM: C[M,N] = epi(A[M,K] @ B[K,N] + bias)
//   EPI: 0 = bias only, 1 = bias+relu, 2 = bias+relu+batchnorm(eval)
//   BTRANS: B stored as [N,K] (used for q @ k_tab^T)
//   Workgroup = 256 threads (8 waves). Tile = 128(M) x 128(N), K step 32.
// ---------------------------------------------------------------------------
template <int EPI, bool BTRANS>
__global__ __launch_bounds__(256) void wmma_gemm(
    const float* __restrict__ A, const float* __restrict__ B,
    float* __restrict__ C, const float* __restrict__ bias,
    const float* __restrict__ bn_g, const float* __restrict__ bn_b,
    const float* __restrict__ bn_m, const float* __restrict__ bn_v,
    int M, int N, int K, int lda, int ldb, int ldc,
    long batchA, long batchB, long batchC) {
  A += (long)blockIdx.z * batchA;
  B += (long)blockIdx.z * batchB;
  C += (long)blockIdx.z * batchC;

  const int m0 = blockIdx.y * 128;
  const int n0 = blockIdx.x * 128;

  // double-buffered B tile, fragment-major: frag(ns,lane) = 32 contiguous bytes
  __shared__ v8bf sBf[2][512];   // 2 x 8 KB

  const int tid  = threadIdx.x;
  const int lane = tid & 31;
  const int wave = tid >> 5;

  // A fragment addressing (ISA 7.12.2 16-bit A layout): lane owns row
  // (lane&15) of the wave's strip; K runs kbA+{0..7}, kbA+16+{0..7}.
  const int arow   = m0 + wave * 16 + (lane & 15);
  const int arow_c = (arow < M) ? arow : (M - 1);          // clamp, no zeroing needed
  const int kbA    = (lane >> 4) * 8;
  const bool lda_vec = ((lda & 3) == 0);
  const bool ldb_vec = ((ldb & 3) == 0);

  // B staging: thread (wave,lane) produces fragment (ns=wave, lane):
  // column cB, K runs kbB+{0..7}, kbB+16+{0..7}.
  const int cB   = n0 + wave * 16 + (lane & 15);
  const int cB_c = (cB < N) ? cB : (N - 1);                // clamp only
  const int kbB  = (lane >> 4) * 8;

  auto stageB = [&](int kk, int buf) {
    v8bf blo, bhi;
    if (kk + 32 <= K) {              // uniform full-K fast path, no selects
      if (BTRANS && ldb_vec) {
        const v4f* vp = (const v4f*)(B + (long)cB_c * ldb + kk + kbB);
        const v4f q0 = vp[0], q1 = vp[1], q2 = vp[4], q3 = vp[5];
        blo = __builtin_shufflevector(cvt4(q0), cvt4(q1), 0, 1, 2, 3, 4, 5, 6, 7);
        bhi = __builtin_shufflevector(cvt4(q2), cvt4(q3), 0, 1, 2, 3, 4, 5, 6, 7);
      } else {
#pragma unroll
        for (int e = 0; e < 16; e += 2) {
          const long k = kk + kbB + ((e < 8) ? e : (e + 8));
          const float v0 = BTRANS ? B[(long)cB_c * ldb + k] : B[k * ldb + cB_c];
          const float v1 = BTRANS ? B[(long)cB_c * ldb + k + 1] : B[(k + 1) * ldb + cB_c];
          const v2bf p = pk2(v0, v1);
          if (e < 8) { blo[e] = p[0]; blo[e + 1] = p[1]; }
          else       { bhi[e - 8] = p[0]; bhi[e - 7] = p[1]; }
        }
      }
    } else {                         // K tail: clamp address, zero-select
#pragma unroll
      for (int e = 0; e < 16; e += 2) {
        const int k0 = kk + kbB + ((e < 8) ? e : (e + 8)), k1 = k0 + 1;
        const int kc0 = (k0 < K) ? k0 : (K - 1);
        const int kc1 = (k1 < K) ? k1 : (K - 1);
        float v0 = BTRANS ? B[(long)cB_c * ldb + kc0] : B[(long)kc0 * ldb + cB_c];
        float v1 = BTRANS ? B[(long)cB_c * ldb + kc1] : B[(long)kc1 * ldb + cB_c];
        v0 = (k0 < K) ? v0 : 0.f;
        v1 = (k1 < K) ? v1 : 0.f;
        const v2bf p = pk2(v0, v1);
        if (e < 8) { blo[e] = p[0]; blo[e + 1] = p[1]; }
        else       { bhi[e - 8] = p[0]; bhi[e - 7] = p[1]; }
      }
    }
    sBf[buf][tid * 2]     = blo;
    sBf[buf][tid * 2 + 1] = bhi;
  };

  auto loadA = [&](int kk) -> v16bf {
    const float* pa = A + (long)arow_c * lda + kk + kbA;
    if ((kk + 32) <= K && lda_vec) { // uniform fast path: 4 aligned v4f loads
      const v4f* vp0 = (const v4f*)pa;
      const v4f* vp1 = (const v4f*)(pa + 16);
      const v4f q0 = vp0[0], q1 = vp0[1], q2 = vp1[0], q3 = vp1[1];
      __builtin_prefetch(pa + 32, 0, 1);   // next K tile -> global_prefetch_b8
      const v8bf lo = __builtin_shufflevector(cvt4(q0), cvt4(q1), 0, 1, 2, 3, 4, 5, 6, 7);
      const v8bf hi = __builtin_shufflevector(cvt4(q2), cvt4(q3), 0, 1, 2, 3, 4, 5, 6, 7);
      return __builtin_shufflevector(lo, hi, 0, 1, 2, 3, 4, 5, 6, 7,
                                     8, 9, 10, 11, 12, 13, 14, 15);
    }
    // K tail: clamp address, zero-select (pairs share the e<8 / e>=8 half)
    v16bf af;
#pragma unroll
    for (int e = 0; e < 16; e += 2) {
      const int k0 = kk + kbA + ((e < 8) ? e : (e + 8));
      const int k1 = k0 + 1;
      const int kc0 = (k0 < K) ? k0 : (K - 1);
      const int kc1 = (k1 < K) ? k1 : (K - 1);
      float v0 = A[(long)arow_c * lda + kc0];
      float v1 = A[(long)arow_c * lda + kc1];
      v0 = (k0 < K) ? v0 : 0.f;
      v1 = (k1 < K) ? v1 : 0.f;
      const v2bf p = pk2(v0, v1);
      af[e] = p[0];
      af[e + 1] = p[1];
    }
    return af;
  };

  v8f acc[8];
#pragma unroll
  for (int i = 0; i < 8; ++i) {
    v8f z = {0.f, 0.f, 0.f, 0.f, 0.f, 0.f, 0.f, 0.f};
    acc[i] = z;
  }

  stageB(0, 0);
  int cur = 0;
  for (int kk = 0; kk < K; kk += 32) {
    const v16bf af = loadA(kk);
    __syncthreads();                         // staging of buf[cur] complete
    if (kk + 32 < K) stageB(kk + 32, cur ^ 1);  // overlap staging with WMMA

    // load all 8 B fragments first (independent regs -> clauseable ds loads),
    // then run the 8 WMMAs back-to-back.
    v8bf flo[8], fhi[8];
#pragma unroll
    for (int ns = 0; ns < 8; ++ns) {
      flo[ns] = sBf[cur][(ns * 32 + lane) * 2];
      fhi[ns] = sBf[cur][(ns * 32 + lane) * 2 + 1];
    }
#pragma unroll
    for (int ns = 0; ns < 8; ++ns) {
      const v16bf bfr = __builtin_shufflevector(
          flo[ns], fhi[ns], 0, 1, 2, 3, 4, 5, 6, 7, 8, 9, 10, 11, 12, 13, 14, 15);
      acc[ns] = __builtin_amdgcn_wmma_f32_16x16x32_bf16(
          false, af, false, bfr, (short)0, acc[ns], false, false);
    }
    cur ^= 1;
  }

  // ---- epilogue: C layout (VGPR r: lanes0-15 -> M=r, lanes16-31 -> M=r+8) ----
  const int rbase  = m0 + wave * 16 + ((lane >= 16) ? 8 : 0);
  const int col_in = lane & 15;
#pragma unroll
  for (int ns = 0; ns < 8; ++ns) {
    const int col = n0 + ns * 16 + col_in;
    if (col >= N) continue;
    const float bi = bias ? bias[col] : 0.f;
    float sc = 1.f, sh = 0.f;
    if (EPI == 2) {
      const float rs = rsqrtf(bn_v[col] + 1e-5f);
      sc = bn_g[col] * rs;
      sh = bn_b[col] - bn_m[col] * sc;
    }
#pragma unroll
    for (int r = 0; r < 8; ++r) {
      const int row = rbase + r;
      if (row < M) {
        float v = acc[ns][r] + bi;
        if (EPI >= 1) v = fmaxf(v, 0.f);
        if (EPI == 2) v = v * sc + sh;
        C[(long)row * ldc + col] = v;
      }
    }
  }
}

// ---------------------------------------------------------------------------
// GIN aggregation, deterministic: out[node] = h[node] + sum_{e: dst==node} h[src[e]]
// One block per node; the node's graph has exactly 256 edges staged in LDS.
// Output stride Fpad (>=F), padded columns zeroed (keeps GEMM A loads aligned).
// ---------------------------------------------------------------------------
__global__ __launch_bounds__(128) void k_gin_agg(
    const float* __restrict__ h, const int* __restrict__ esrc,
    const int* __restrict__ edst, float* __restrict__ out, int F, int Fpad) {
  const int node = blockIdx.x;
  const int g = node >> 6;  // 64 nodes per graph
  __shared__ int s_src[256];
  __shared__ int s_dst[256];
  for (int e = threadIdx.x; e < 256; e += blockDim.x) {
    s_src[e] = esrc[g * 256 + e];
    s_dst[e] = edst[g * 256 + e];
  }
  __syncthreads();
  for (int f = threadIdx.x; f < Fpad; f += blockDim.x) {
    if (f < F) {
      float acc = h[(long)node * F + f];
      for (int e = 0; e < 256; ++e)
        if (s_dst[e] == node) acc += h[(long)s_src[e] * F + f];
      out[(long)node * Fpad + f] = acc;
    } else {
      out[(long)node * Fpad + f] = 0.f;
    }
  }
}

// per-graph vocab histogram of target[b, 0:1000]
__global__ __launch_bounds__(256) void k_hist(const int* __restrict__ target,
                                              int* __restrict__ cnt) {
  __shared__ int c[32];
  const int b = blockIdx.x;
  if (threadIdx.x < 32) c[threadIdx.x] = 0;
  __syncthreads();
  for (int l = threadIdx.x; l < 1000; l += blockDim.x)
    atomicAdd(&c[target[b * 1000 + l]], 1);
  __syncthreads();
  if (threadIdx.x < 32) cnt[b * 32 + threadIdx.x] = c[threadIdx.x];
}

// count-weighted softmax * V: o[row,:] from s_tab[row, 0:26], cnt[b,:], v_tab
__global__ __launch_bounds__(128) void k_attn(
    const float* __restrict__ s_tab, const int* __restrict__ cnt,
    const float* __restrict__ v_tab, float* __restrict__ o) {
  const int row = blockIdx.x;       // node index
  const int d   = threadIdx.x;      // 0..127
  const int b   = row >> 6;
  const float scale = 0.08838834764831845f;  // 1/sqrt(128)
  __shared__ float sh_s[26];
  __shared__ int   sh_c[26];
  if (threadIdx.x < 26) {
    sh_s[threadIdx.x] = s_tab[(long)row * 32 + threadIdx.x] * scale;
    sh_c[threadIdx.x] = cnt[b * 32 + threadIdx.x];
  }
  __syncthreads();
  float m = -1e30f;
#pragma unroll
  for (int v = 0; v < 26; ++v)
    if (sh_c[v] > 0) m = fmaxf(m, sh_s[v]);
  float den = 0.f, num = 0.f;
#pragma unroll
  for (int v = 0; v < 26; ++v) {
    const int c = sh_c[v];
    if (c > 0) {
      const float e = __expf(sh_s[v] - m) * (float)c;
      den += e;
      num += e * v_tab[v * 128 + d];
    }
  }
  o[(long)row * 128 + d] = num / den;
}

// global_add_pool over the 64 nodes of each graph
__global__ __launch_bounds__(128) void k_pool(const float* __restrict__ a,
                                              float* __restrict__ pooled) {
  const int b = blockIdx.x, d = threadIdx.x;
  float s = 0.f;
  for (int n = 0; n < 64; ++n) s += a[((long)b * 64 + n) * 128 + d];
  pooled[b * 128 + d] = s;
}

// xt[b,j] = b_xt[j] + sum_l pos_tab[l, target[b,l], j]  -> xc[:, 128:256]
__global__ __launch_bounds__(128) void k_xt(
    const float* __restrict__ pos_tab, const int* __restrict__ target,
    const float* __restrict__ b_xt, float* __restrict__ xc) {
  const int b = blockIdx.x, j = threadIdx.x;
  __shared__ int tg[1000];
  for (int l = threadIdx.x; l < 1000; l += blockDim.x) tg[l] = target[b * 1000 + l];
  __syncthreads();
  float s = b_xt[j];
  for (int l = 0; l < 1000; ++l) s += pos_tab[(long)l * 4096 + tg[l] * 128 + j];
  xc[b * 256 + 128 + j] = s;
}

// out[b] = b_out + f2[b,:] . w_out
__global__ __launch_bounds__(256) void k_out(const float* __restrict__ f2,
                                             const float* __restrict__ w,
                                             const float* __restrict__ bo,
                                             float* __restrict__ out) {
  const int b = blockIdx.x * blockDim.x + threadIdx.x;
  if (b < 256) {
    float s = bo[0];
    for (int j = 0; j < 256; ++j) s += f2[b * 256 + j] * w[j];
    out[b] = s;
  }
}

// ---------------------------------------------------------------------------
// host-side dispatch
// ---------------------------------------------------------------------------
static inline void gemm(hipStream_t st, int epi, bool btrans, const float* A,
                        const float* B, float* C, const float* bias,
                        const float* g, const float* be, const float* mn,
                        const float* vr, int M, int N, int K, int lda, int ldb,
                        int ldc, int batches = 1, long bA = 0, long bB = 0,
                        long bC = 0) {
  dim3 grid((N + 127) / 128, (M + 127) / 128, batches), blk(256);
  if (btrans)
    wmma_gemm<0, true><<<grid, blk, 0, st>>>(A, B, C, bias, g, be, mn, vr, M, N, K, lda, ldb, ldc, bA, bB, bC);
  else if (epi == 0)
    wmma_gemm<0, false><<<grid, blk, 0, st>>>(A, B, C, bias, g, be, mn, vr, M, N, K, lda, ldb, ldc, bA, bB, bC);
  else if (epi == 1)
    wmma_gemm<1, false><<<grid, blk, 0, st>>>(A, B, C, bias, g, be, mn, vr, M, N, K, lda, ldb, ldc, bA, bB, bC);
  else
    wmma_gemm<2, false><<<grid, blk, 0, st>>>(A, B, C, bias, g, be, mn, vr, M, N, K, lda, ldb, ldc, bA, bB, bC);
}

extern "C" void kernel_launch(void* const* d_in, const int* in_sizes, int n_in,
                              void* d_out, int out_size, void* d_ws,
                              size_t ws_size, hipStream_t stream) {
  (void)in_sizes; (void)n_in; (void)out_size; (void)ws_size;
  // inputs, setup_inputs() insertion order (nested dicts depth-first)
  const float* x    = (const float*)d_in[0];
  const float* g1w1 = (const float*)d_in[1];  const float* g1b1 = (const float*)d_in[2];
  const float* g1w2 = (const float*)d_in[3];  const float* g1b2 = (const float*)d_in[4];
  const float* bn1g = (const float*)d_in[5];  const float* bn1b = (const float*)d_in[6];
  const float* bn1m = (const float*)d_in[7];  const float* bn1v = (const float*)d_in[8];
  const float* g2w1 = (const float*)d_in[9];  const float* g2b1 = (const float*)d_in[10];
  const float* g2w2 = (const float*)d_in[11]; const float* g2b2 = (const float*)d_in[12];
  const float* bn2g = (const float*)d_in[13]; const float* bn2b = (const float*)d_in[14];
  const float* bn2m = (const float*)d_in[15]; const float* bn2v = (const float*)d_in[16];
  const float* g3w1 = (const float*)d_in[17]; const float* g3b1 = (const float*)d_in[18];
  const float* g3w2 = (const float*)d_in[19]; const float* g3b2 = (const float*)d_in[20];
  const float* bn3g = (const float*)d_in[21]; const float* bn3b = (const float*)d_in[22];
  const float* bn3m = (const float*)d_in[23]; const float* bn3v = (const float*)d_in[24];
  const float* emb  = (const float*)d_in[25];
  const float* wq   = (const float*)d_in[26]; const float* bq   = (const float*)d_in[27];
  const float* wk   = (const float*)d_in[28]; const float* bk   = (const float*)d_in[29];
  const float* wv   = (const float*)d_in[30]; const float* bv   = (const float*)d_in[31];
  const float* wo   = (const float*)d_in[32]; const float* bo   = (const float*)d_in[33];
  const float* wxd  = (const float*)d_in[34]; const float* bxd  = (const float*)d_in[35];
  const float* wxt  = (const float*)d_in[36]; const float* bxt  = (const float*)d_in[37];
  const float* wfc1 = (const float*)d_in[38]; const float* bfc1 = (const float*)d_in[39];
  const float* wfc2 = (const float*)d_in[40]; const float* bfc2 = (const float*)d_in[41];
  const float* wout = (const float*)d_in[42]; const float* bout = (const float*)d_in[43];
  const int* esrc   = (const int*)d_in[44];
  const int* edst   = (const int*)d_in[45];
  // d_in[46] = batch (implicit in layout)
  const int* target = (const int*)d_in[47];

  const int NTOT = 16384;

  // workspace carve-up (floats)
  float* ws     = (float*)d_ws;
  float* sum    = ws;                   // 16384*128 (layer1 uses stride 80)
  float* tbuf   = sum + 2097152;        // 16384*128
  float* hbuf   = tbuf + 2097152;       // 16384*128
  float* ktab   = hbuf + 2097152;       // 32*128
  float* vtab   = ktab + 4096;          // 32*128
  float* stab   = vtab + 4096;          // 16384*32
  int*   cnt    = (int*)(stab + 524288);// 256*32
  float* pooled = (float*)(cnt + 8192); // 256*128
  float* xc     = pooled + 32768;       // 256*256
  float* postab = xc + 65536;           // 1000*32*128
  float* f1     = postab + 4096000;     // 256*1024
  float* f2     = f1 + 262144;          // 256*256

  // ---- GIN layer 1 (F_IN=78, padded to stride 80 for aligned A loads) ----
  k_gin_agg<<<NTOT, 128, 0, stream>>>(x, esrc, edst, sum, 78, 80);
  gemm(stream, 1, false, sum, g1w1, tbuf, g1b1, 0, 0, 0, 0, NTOT, 128, 78, 80, 128, 128);
  gemm(stream, 2, false, tbuf, g1w2, hbuf, g1b2, bn1g, bn1b, bn1m, bn1v, NTOT, 128, 128, 128, 128, 128);
  // ---- GIN layer 2 ----
  k_gin_agg<<<NTOT, 128, 0, stream>>>(hbuf, esrc, edst, sum, 128, 128);
  gemm(stream, 1, false, sum, g2w1, tbuf, g2b1, 0, 0, 0, 0, NTOT, 128, 128, 128, 128, 128);
  gemm(stream, 2, false, tbuf, g2w2, hbuf, g2b2, bn2g, bn2b, bn2m, bn2v, NTOT, 128, 128, 128, 128, 128);
  // ---- GIN layer 3 ----
  k_gin_agg<<<NTOT, 128, 0, stream>>>(hbuf, esrc, edst, sum, 128, 128);
  gemm(stream, 1, false, sum, g3w1, tbuf, g3b1, 0, 0, 0, 0, NTOT, 128, 128, 128, 128, 128);
  gemm(stream, 2, false, tbuf, g3w2, hbuf, g3b2, bn3g, bn3b, bn3m, bn3v, NTOT, 128, 128, 128, 128, 128);

  // ---- attention (26-vocab factorization) ----
  gemm(stream, 0, false, hbuf, wq, tbuf, bq, 0, 0, 0, 0, NTOT, 128, 128, 128, 128, 128); // q -> tbuf
  gemm(stream, 0, false, emb, wk, ktab, bk, 0, 0, 0, 0, 26, 128, 128, 128, 128, 128);    // k table
  gemm(stream, 0, false, emb, wv, vtab, bv, 0, 0, 0, 0, 26, 128, 128, 128, 128, 128);    // v table
  gemm(stream, 0, true,  tbuf, ktab, stab, nullptr, 0, 0, 0, 0, NTOT, 26, 128, 128, 128, 32); // q @ kT
  k_hist<<<256, 256, 0, stream>>>(target, cnt);
  k_attn<<<NTOT, 128, 0, stream>>>(stab, cnt, vtab, sum);                                // o -> sum
  gemm(stream, 0, false, sum, wo, hbuf, bo, 0, 0, 0, 0, NTOT, 128, 128, 128, 128, 128);  // o @ wo

  // ---- pooling + xd branch (writes xc[:, 0:128]) ----
  k_pool<<<256, 128, 0, stream>>>(hbuf, pooled);
  gemm(stream, 1, false, pooled, wxd, xc, bxd, 0, 0, 0, 0, 256, 128, 128, 128, 128, 256);

  // ---- xt branch: batched pos_tab GEMM (1000 batches) + gather (xc[:,128:256]) ----
  gemm(stream, 0, false, emb, wxt, postab, nullptr, 0, 0, 0, 0, 26, 128, 128, 128, 128, 128,
       1000, 0, 16384, 4096);
  k_xt<<<256, 128, 0, stream>>>(postab, target, bxt, xc);

  // ---- head MLP ----
  gemm(stream, 1, false, xc, wfc1, f1, bfc1, 0, 0, 0, 0, 256, 1024, 256, 256, 1024, 1024);
  gemm(stream, 1, false, f1, wfc2, f2, bfc2, 0, 0, 0, 0, 256, 256, 1024, 1024, 256, 256);
  k_out<<<1, 256, 0, stream>>>(f2, wout, bout, (float*)d_out);
}
